// GINEncoder_14912126452117
// MI455X (gfx1250) — matrix operand order repaired and checked by reading, not compiled, not run on hardware
//
#include <hip/hip_runtime.h>

#define NODES   100000
#define EDGES   1600000
#define GRAPHS  64
#define HID     128
#define ODIM    256
#define LAYERS  5

typedef __attribute__((ext_vector_type(16))) __bf16        v16bf;
typedef __attribute__((ext_vector_type(8)))  float         v8f;
typedef __attribute__((ext_vector_type(8)))  unsigned int  v8u;

// ---- fp32 <-> bf16 helpers (round-to-nearest-even, bit level) ----
__device__ __forceinline__ unsigned short f2bf(float x) {
  unsigned int u = __builtin_bit_cast(unsigned int, x);
  u += 0x7fffu + ((u >> 16) & 1u);
  return (unsigned short)(u >> 16);
}
__device__ __forceinline__ float bf2f(unsigned short h) {
  unsigned int u = ((unsigned int)h) << 16;
  return __builtin_bit_cast(float, u);
}
// split x into bf16 hi/lo, return packed pair (x0 in low half, x1 in high half)
__device__ __forceinline__ void split2(float x0, float x1, unsigned int& hp, unsigned int& lp) {
  unsigned short h0 = f2bf(x0), h1 = f2bf(x1);
  unsigned short l0 = f2bf(x0 - bf2f(h0)), l1 = f2bf(x1 - bf2f(h1));
  hp = (unsigned int)h0 | ((unsigned int)h1 << 16);
  lp = (unsigned int)l0 | ((unsigned int)l1 << 16);
}

// ---- simple utility kernels ----
__global__ void copy_f4(const float4* __restrict__ a, float4* __restrict__ b, int n4) {
  int i = blockIdx.x * blockDim.x + threadIdx.x;
  if (i < n4) b[i] = a[i];
}
__global__ void zero_f(float* __restrict__ p, int n) {
  int i = blockIdx.x * blockDim.x + threadIdx.x;
  if (i < n) p[i] = 0.f;
}

// ---- edge aggregation: z += h[src] scattered to dst (z pre-initialized to h) ----
// one wave32 per edge, 4 floats per lane (float4 coalesced gather + f32 atomics)
__global__ __launch_bounds__(256) void scatter_add_k(
    const float* __restrict__ h, const int* __restrict__ src,
    const int* __restrict__ dst, float* __restrict__ z) {
  long long tid = (long long)blockIdx.x * blockDim.x + threadIdx.x;
  int e = (int)(tid >> 5);
  int l = (int)(tid & 31);
  if (e >= EDGES) return;
  int s = src[e], d = dst[e];
  const float4 v = *(const float4*)(h + (size_t)s * HID + l * 4);
  float* p = z + (size_t)d * HID + l * 4;
  atomicAdd(p + 0, v.x);
  atomicAdd(p + 1, v.y);
  atomicAdd(p + 2, v.z);
  atomicAdd(p + 3, v.w);
}

// ---- split fp32 weights into bf16 hi/lo, emitted in per-lane WMMA B-fragment order ----
// B fragment (32x16, K x N) for (ks, nt, lane): n = nt*16 + lane%16,
// uint j holds K-pair ks*32 + (j>>2)*16 + (lane>>4)*8 + 2(j&3) .. +1.
// Global layout: frag[((ks*8 + nt)*32 + lane)*8 + j]   (8192 uints = 32KB per matrix)
__global__ void split_w_frag(const float* __restrict__ w,
                             unsigned int* __restrict__ hfrag,
                             unsigned int* __restrict__ lfrag) {
  int idx = blockIdx.x * blockDim.x + threadIdx.x;   // 8192
  if (idx >= 8192) return;
  int j    = idx & 7;
  int lane = (idx >> 3) & 31;
  int nt   = (idx >> 8) & 7;
  int ks   = idx >> 11;
  int n = nt * 16 + (lane & 15);
  int k = ks * 32 + (j >> 2) * 16 + (lane >> 4) * 8 + 2 * (j & 3);
  unsigned int hp, lp;
  split2(w[k * HID + n], w[(k + 1) * HID + n], hp, lp);
  hfrag[idx] = hp;
  lfrag[idx] = lp;
}

// ---- GEMM: C[M,128] = A[M,128] @ W[128,128] + bias ----
// MODE 0: A raw.  MODE 1: A = relu(A*scale[col]+shift[col]) (fused BN+ReLU prologue).
// Epilogue fuses per-column sum/sumsq for the following BatchNorm (atomics into stats).
// bf16 split precision: A*B ~= Ah*Bh + Ah*Bl + Al*Bh, fp32 accumulate.
// LDS holds A and B in per-lane fragment order -> each fragment = 2x ds_load_b128.
// Block = 256 thr (8 waves), tile = 128 rows x 128 cols; wave -> 16 rows x 8 n-tiles.
template <int MODE>
__global__ __launch_bounds__(256) void gin_gemm(
    const float* __restrict__ gA, const unsigned int* __restrict__ whf,
    const unsigned int* __restrict__ wlf, const float* __restrict__ bias,
    const float* __restrict__ scale, const float* __restrict__ shift,
    float* __restrict__ gC, float* __restrict__ stats, int M) {
  extern __shared__ char smem[];
  unsigned int* AhL = (unsigned int*)smem;   // 8192 uints (32KB), A hi fragments
  unsigned int* AlL = AhL + 8192;            // A lo fragments
  unsigned int* BhL = AlL + 8192;            // B hi fragments
  unsigned int* BlL = BhL + 8192;            // B lo fragments

  const int m0 = blockIdx.x * 128;
  const int tid = threadIdx.x;

  // stage A tile: read float2 K-pairs, transform, split, store in fragment order.
  // A fragment (16x32): uint j<4 -> K = ks*32 + khalf*8 + 2j ; j>=4 -> +16.
  // For element kk (0..31): khalf=(kk>>3)&1, j=((kk&16)>>2)+((kk&7)>>1).
  for (int idx = tid; idx < 8192; idx += 256) {
    int row = idx >> 6;          // 0..127
    int k   = (idx & 63) * 2;    // even K
    int grow = m0 + row;
    float x0 = 0.f, x1 = 0.f;
    if (grow < M) {
      const float2 v = *(const float2*)(gA + (size_t)grow * HID + k);
      x0 = v.x; x1 = v.y;
      if (MODE == 1) {
        x0 = fmaxf(x0 * scale[k] + shift[k], 0.f);
        x1 = fmaxf(x1 * scale[k + 1] + shift[k + 1], 0.f);
      }
    }
    unsigned int hp, lp;
    split2(x0, x1, hp, lp);
    int ks = k >> 5, kk = k & 31;
    int khalf = (kk >> 3) & 1;
    int j = ((kk & 16) >> 2) + ((kk & 7) >> 1);
    int ui = ((row * 4 + ks) * 2 + khalf) * 8 + j;
    AhL[ui] = hp;
    AlL[ui] = lp;
  }
  // stage weights: already in fragment order in global, linear b128 copy
  {
    const uint4* gh = (const uint4*)whf;
    const uint4* gl = (const uint4*)wlf;
    uint4* sh = (uint4*)BhL;
    uint4* sl = (uint4*)BlL;
    for (int idx = tid; idx < 2048; idx += 256) {
      sh[idx] = gh[idx];
      sl[idx] = gl[idx];
    }
  }
  __syncthreads();

  const int wave  = tid >> 5;
  const int lane  = tid & 31;
  const int nrow  = lane & 15;
  const int khalf = lane >> 4;
  const int mrow  = (wave << 4) + nrow;   // A row within tile

  v8f zero8 = {0.f, 0.f, 0.f, 0.f, 0.f, 0.f, 0.f, 0.f};
  v8f acc[8];
#pragma unroll
  for (int i = 0; i < 8; ++i) acc[i] = zero8;

#pragma unroll
  for (int ks = 0; ks < 4; ++ks) {        // K = 128 in 4 steps of 32
    const int abase = ((mrow * 4 + ks) * 2 + khalf) * 8;
    v16bf ah = __builtin_bit_cast(v16bf, *(const v8u*)(AhL + abase));
    v16bf al = __builtin_bit_cast(v16bf, *(const v8u*)(AlL + abase));
#pragma unroll
    for (int nt = 0; nt < 8; ++nt) {
      const int bbase = ((ks * 8 + nt) * 32 + lane) * 8;
      v16bf bh = __builtin_bit_cast(v16bf, *(const v8u*)(BhL + bbase));
      v16bf bl = __builtin_bit_cast(v16bf, *(const v8u*)(BlL + bbase));
      acc[nt] = __builtin_amdgcn_wmma_f32_16x16x32_bf16(
          false, ah, false, bh, (short)0, acc[nt], false, false);
      acc[nt] = __builtin_amdgcn_wmma_f32_16x16x32_bf16(
          false, ah, false, bl, (short)0, acc[nt], false, false);
      acc[nt] = __builtin_amdgcn_wmma_f32_16x16x32_bf16(
          false, al, false, bh, (short)0, acc[nt], false, false);
    }
  }

  // epilogue: +bias, store, fused column sum/sumsq for BatchNorm
  const int rowbase = m0 + (wave << 4) + khalf * 8;  // C layout: VGPR r -> M=khalf*8+r
#pragma unroll
  for (int nt = 0; nt < 8; ++nt) {
    const int c = nt * 16 + nrow;
    const float b = bias[c];
    float s = 0.f, s2 = 0.f;
#pragma unroll
    for (int r = 0; r < 8; ++r) {
      int row = rowbase + r;
      float v = acc[nt][r] + b;
      if (row < M) {
        gC[(size_t)row * HID + c] = v;
        s += v;
        s2 += v * v;
      }
    }
    // pair lanes (L, L+16) cover the same column
    s  += __shfl_xor(s, 16, 32);
    s2 += __shfl_xor(s2, 16, 32);
    if (lane < 16) {
      atomicAdd(&stats[c], s);
      atomicAdd(&stats[HID + c], s2);
    }
  }
}

// ---- finalize BN: scale/shift from accumulated sum/sumsq; reset sums ----
__global__ void bn_fin(float* __restrict__ stats, const float* __restrict__ gamma,
                       const float* __restrict__ beta, float invN) {
  int c = threadIdx.x;
  float mean = stats[c] * invN;
  float var  = stats[HID + c] * invN - mean * mean;
  float rstd = rsqrtf(var + 1e-5f);
  float sc = gamma[c] * rstd;
  stats[2 * HID + c] = sc;
  stats[3 * HID + c] = beta[c] - mean * sc;
  stats[c] = 0.f;
  stats[HID + c] = 0.f;
}

// ---- residual: h += relu(t*scale2 + shift2) ----
__global__ void residual_k(float* __restrict__ h, const float* __restrict__ t,
                           const float* __restrict__ stats) {
  int i = blockIdx.x * blockDim.x + threadIdx.x;
  if (i >= NODES * HID) return;
  int c = i & (HID - 1);
  h[i] += fmaxf(t[i] * stats[2 * HID + c] + stats[3 * HID + c], 0.f);
}

// ---- per-graph mean pool (sums + counts via atomics) ----
__global__ __launch_bounds__(256) void pool_k(const float* __restrict__ h,
                                              const int* __restrict__ batch,
                                              float* __restrict__ pool) {
  long long tid = (long long)blockIdx.x * blockDim.x + threadIdx.x;
  int n = (int)(tid >> 5);
  int l = (int)(tid & 31);
  if (n >= NODES) return;
  int g = batch[n];
  const float4 v = *(const float4*)(h + (size_t)n * HID + l * 4);
  float* p = pool + (size_t)g * HID + l * 4;
  atomicAdd(p + 0, v.x);
  atomicAdd(p + 1, v.y);
  atomicAdd(p + 2, v.z);
  atomicAdd(p + 3, v.w);
  if (l == 0) atomicAdd(pool + GRAPHS * HID + g, 1.f);
}

// ---- graph_emb = (poolsum/count) @ wp + bp   (64x256, tiny) ----
__global__ void proj_k(const float* __restrict__ pool, const float* __restrict__ wp,
                       const float* __restrict__ bp, float* __restrict__ out) {
  int tid = blockIdx.x * blockDim.x + threadIdx.x;   // 16384
  int g = tid >> 8, o = tid & 255;
  float inv = 1.f / fmaxf(pool[GRAPHS * HID + g], 1.f);
  float a = bp[o];
  for (int k = 0; k < HID; ++k)
    a += pool[g * HID + k] * inv * wp[k * ODIM + o];
  out[tid] = a;
}

extern "C" void kernel_launch(void* const* d_in, const int* in_sizes, int n_in,
                              void* d_out, int out_size, void* d_ws, size_t ws_size,
                              hipStream_t stream) {
  const float* x        = (const float*)d_in[0];
  const int*   ei       = (const int*)d_in[1];
  const int*   batch    = (const int*)d_in[2];
  const float* w1       = (const float*)d_in[3];
  const float* b1       = (const float*)d_in[4];
  const float* gamma1   = (const float*)d_in[5];
  const float* beta1    = (const float*)d_in[6];
  const float* w2       = (const float*)d_in[7];
  const float* b2       = (const float*)d_in[8];
  const float* gamma_bn = (const float*)d_in[9];
  const float* beta_bn  = (const float*)d_in[10];
  const float* wp       = (const float*)d_in[11];
  const float* bp       = (const float*)d_in[12];

  const int* src = ei;
  const int* dst = ei + EDGES;

  char* ws = (char*)d_ws;
  float* z     = (float*)ws;                        // 100000x128 f32 (51.2 MB)
  float* t     = (float*)(ws + 51200000);           // 100000x128 f32 (51.2 MB)
  float* stats = (float*)(ws + 102400000);          // [sum,sumsq,scale,shift] x 128
  float* pool  = (float*)(ws + 102402048);          // 64x128 sums + 64 counts
  unsigned int* w1h = (unsigned int*)(ws + 102435072);   // 8192 uints each
  unsigned int* w1l = w1h + 8192;
  unsigned int* w2h = w1l + 8192;
  unsigned int* w2l = w2h + 8192;

  float* gout = (float*)d_out;           // graph_emb [64x256]
  float* h    = gout + GRAPHS * ODIM;    // node_emb  [100000x128] built in place

  const int n4   = NODES * HID / 4;      // 3,200,000 float4
  const int gcpy = n4 / 256;             // 12500
  const int ggem = (NODES + 127) / 128;  // 782
  const float invN = 1.f / (float)NODES;

  // h = x
  copy_f4<<<gcpy, 256, 0, stream>>>((const float4*)x, (float4*)h, n4);
  // zero BN accumulators once (bn_fin re-zeroes after each use)
  zero_f<<<1, 256, 0, stream>>>(stats, 2 * HID);

  for (int i = 0; i < LAYERS; ++i) {
    // z = h ; z += scatter(h[src] -> dst)   => z = (1+eps)h + agg with eps=0
    copy_f4<<<gcpy, 256, 0, stream>>>((const float4*)h, (float4*)z, n4);
    scatter_add_k<<<EDGES * 32 / 256, 256, 0, stream>>>(h, src, dst, z);

    split_w_frag<<<32, 256, 0, stream>>>(w1 + i * HID * HID, w1h, w1l);
    split_w_frag<<<32, 256, 0, stream>>>(w2 + i * HID * HID, w2h, w2l);

    // t = z @ W1 + b1  (+ column stats for inner BN)
    gin_gemm<0><<<ggem, 256, 131072, stream>>>(z, w1h, w1l, b1 + i * HID,
                                               nullptr, nullptr, t, stats, NODES);
    bn_fin<<<1, 128, 0, stream>>>(stats, gamma1 + i * HID, beta1 + i * HID, invN);

    // t = relu(bn1(t)) @ W2 + b2, in place (+ column stats for outer BN)
    gin_gemm<1><<<ggem, 256, 131072, stream>>>(t, w2h, w2l, b2 + i * HID,
                                               stats + 2 * HID, stats + 3 * HID,
                                               t, stats, NODES);
    bn_fin<<<1, 128, 0, stream>>>(stats, gamma_bn + i * HID, beta_bn + i * HID, invN);

    // h += relu(bn2(t))
    residual_k<<<NODES * HID / 256, 256, 0, stream>>>(h, t, stats);
  }

  // global mean pool + output projection
  zero_f<<<33, 256, 0, stream>>>(pool, GRAPHS * HID + GRAPHS);
  pool_k<<<NODES * 32 / 256, 256, 0, stream>>>(h, batch, pool);
  proj_k<<<GRAPHS, 256, 0, stream>>>(pool, wp, bp, gout);
}
